// GIHyperTransformer_43800076484706
// MI455X (gfx1250) — matrix operand order repaired and checked by reading, not compiled, not run on hardware
//
#include <hip/hip_runtime.h>
#include <math.h>
#include <stdint.h>

// ---------------------------------------------------------------------------
// GIHC hyper-transformer forward for MI455X (gfx1250).
// - All large GEMMs: v_wmma_f32_16x16x32_bf16, f32 accumulate, fused epilogues
// - f32 -> bf16 conversion fused into LDS staging (native cvt)
// - Attention probabilities written as bf16 by softmax; the P@V GEMM stages its
//   A-tiles with GLOBAL_LOAD_ASYNC_TO_LDS_B128 (ASYNCcnt) when available.
// ---------------------------------------------------------------------------

#define BM 128
#define BN 128
#define BKK 32
#define LDSK 40   // BKK + 8 pad (bf16 units) -> 80B row stride, conflict-friendly

#if __has_builtin(__builtin_amdgcn_global_load_async_to_lds_b128)
#define ASYNC_LDS 1
#else
#define ASYNC_LDS 0
#endif

#define ASG __attribute__((address_space(1)))
#define ASL __attribute__((address_space(3)))

typedef __bf16 bhalf;
typedef __attribute__((ext_vector_type(16))) __bf16 v16bf;
typedef __attribute__((ext_vector_type(8)))  float  v8f;
typedef int v4i __attribute__((vector_size(16)));   // matches builtin param type
typedef ASG v4i* gvp;                               // global int4*
typedef ASL v4i* lvp;                               // LDS int4*

struct alignas(16) F4 { float x, y, z, w; };
struct alignas(8)  U2 { unsigned a, b; };

union AFrag { v16bf v; F4 f[2]; };
union CFrag { v8f v; float f[8]; };

__device__ __forceinline__ unsigned short bf_bits(float x) {
  return __builtin_bit_cast(unsigned short, (__bf16)x);
}
__device__ __forceinline__ unsigned pack_bf2(float a, float b) {
  return (unsigned)bf_bits(a) | ((unsigned)bf_bits(b) << 16);
}

__device__ __forceinline__ void wait_async0() {
#if __has_builtin(__builtin_amdgcn_s_wait_asynccnt)
  __builtin_amdgcn_s_wait_asynccnt(0);
#else
  asm volatile("s_wait_asynccnt 0" ::: "memory");
#endif
}

// stage one 128x32 f32 A tile (unless ABF) and one 128x32 B tile into registers
template<bool BT, bool ABF>
__device__ __forceinline__ void stage_load(const float* __restrict__ A, long long lda,
                                           const float* __restrict__ B, long long ldb,
                                           int mBase, int nBase, int M, int N, int kb,
                                           int srow, int scol,
                                           float* __restrict__ areg, float* __restrict__ breg)
{
#pragma unroll
  for (int i = 0; i < 4; ++i) {
    int r = srow + 32 * i;
    if (!ABF) {
      int gm = mBase + r;
      if (gm < M) {
        F4 v = *(const F4*)(A + (long long)gm * lda + kb + scol);
        areg[4*i+0] = v.x; areg[4*i+1] = v.y; areg[4*i+2] = v.z; areg[4*i+3] = v.w;
      } else {
        areg[4*i+0] = 0.f; areg[4*i+1] = 0.f; areg[4*i+2] = 0.f; areg[4*i+3] = 0.f;
      }
    }
    int gn = nBase + r;
    if (!BT) {
      if (gn < N) {
        F4 v = *(const F4*)(B + (long long)gn * ldb + kb + scol);
        breg[4*i+0] = v.x; breg[4*i+1] = v.y; breg[4*i+2] = v.z; breg[4*i+3] = v.w;
      } else {
        breg[4*i+0] = 0.f; breg[4*i+1] = 0.f; breg[4*i+2] = 0.f; breg[4*i+3] = 0.f;
      }
    } else {
      // B element (n,k) lives at B[k*ldb + n]  (K-major B, used for P@V)
#pragma unroll
      for (int j = 0; j < 4; ++j)
        breg[4*i+j] = (gn < N) ? B[(long long)(kb + scol + j) * ldb + gn] : 0.f;
    }
  }
}

// C = epi( alpha * (A @ B^T) + bias[n] ), optionally 2-level batched via blockIdx.z
// EPI: 0 = linear, 1 = exact GELU, 2 = sigmoid, 3 = 2*sigmoid
// ABF: A is already bf16 in memory (lda in elements) -> async copy to LDS
template<int EPI, bool BT, bool ABF>
__global__ __launch_bounds__(256)
void gemm_nt_kernel(const void* __restrict__ Ao, long long lda,
                    const float* __restrict__ B, long long ldb,
                    float* __restrict__ Co, long long ldc,
                    const float* __restrict__ bias,
                    const float* __restrict__ aptr, float aimm,
                    int M, int N, int K, int nH,
                    long long sAb, long long sAh,
                    long long sBb, long long sBh,
                    long long sCb, long long sCh)
{
  __shared__ alignas(16) unsigned short As[BM * LDSK];
  __shared__ alignas(16) unsigned short Bs[BN * LDSK];

  long long aoff;
  {
    int z  = blockIdx.z;
    int zb = z / nH, zh = z % nH;
    aoff = (long long)zb * sAb + (long long)zh * sAh;
    B  += (long long)zb * sBb + (long long)zh * sBh;
    Co += (long long)zb * sCb + (long long)zh * sCh;
  }
  const float* Af = (const float*)Ao + (ABF ? 0 : aoff);
  const unsigned short* Ah = (const unsigned short*)Ao + (ABF ? aoff : 0);

  const int mBase = blockIdx.y * BM;
  const int nBase = blockIdx.x * BN;
  const int t    = threadIdx.x;
  const int lane = t & 31;
  const int wave = t >> 5;
  const int hlf  = lane >> 4;         // which 16-lane half
  const int l16  = lane & 15;
  const int wm   = (wave >> 2) * 64;  // wave M offset inside block tile (2 waves)
  const int wn   = (wave & 3) * 32;   // wave N offset inside block tile (4 waves)

  const int srow = t >> 3;            // 0..31 staging row
  const int scol = (t & 7) * 4;       // 0..28 staging col (float4)

  float areg[16], breg[16];
  CFrag acc[4][2];
#pragma unroll
  for (int i = 0; i < 4; ++i)
#pragma unroll
    for (int j = 0; j < 2; ++j)
#pragma unroll
      for (int r = 0; r < 8; ++r) acc[i][j].f[r] = 0.f;

  const int kIters = K / BKK;   // all K in this model are multiples of 32

  stage_load<BT, ABF>(Af, lda, B, ldb, mBase, nBase, M, N, 0, srow, scol, areg, breg);

  for (int kt = 0; kt < kIters; ++kt) {
    __syncthreads();
    if (ABF) {
      // copy bf16 A tile straight into LDS: 128 rows x 64B, 2 x 16B per thread
      const int row  = t >> 1;
      const int gm   = mBase + row;
      const int segb = (t & 1) * 2;
      const int kb   = kt * BKK;
#pragma unroll
      for (int s2 = 0; s2 < 2; ++s2) {
        const int s = segb + s2;
        unsigned short* dst = &As[row * LDSK + s * 8];
        if (gm < M) {
          const unsigned short* src = Ah + (long long)gm * lda + kb + s * 8;
#if ASYNC_LDS
          __builtin_amdgcn_global_load_async_to_lds_b128(
              (gvp)(size_t)src, (lvp)(unsigned)(size_t)dst, 0, 0);
#else
          *(F4*)dst = *(const F4*)src;
#endif
        } else {
          ((U2*)dst)[0] = U2{0u, 0u};
          ((U2*)dst)[1] = U2{0u, 0u};
        }
      }
    }
    // registers -> LDS (f32 -> bf16 RNE, packed dword stores)
#pragma unroll
    for (int i = 0; i < 4; ++i) {
      int r = srow + 32 * i;
      if (!ABF)
        *(U2*)&As[r * LDSK + scol] =
            U2{pack_bf2(areg[4*i+0], areg[4*i+1]), pack_bf2(areg[4*i+2], areg[4*i+3])};
      *(U2*)&Bs[r * LDSK + scol] =
          U2{pack_bf2(breg[4*i+0], breg[4*i+1]), pack_bf2(breg[4*i+2], breg[4*i+3])};
    }
#if ASYNC_LDS
    if (ABF) wait_async0();
#endif
    __syncthreads();
    if (kt + 1 < kIters)
      stage_load<BT, ABF>(Af, lda, B, ldb, mBase, nBase, M, N, (kt + 1) * BKK,
                          srow, scol, areg, breg);

    // fragment loads per ISA 16-bit A(16x32)/B(32x16) wave32 layouts
    AFrag af[4], bfr[2];
#pragma unroll
    for (int mt = 0; mt < 4; ++mt) {
      int row = wm + mt * 16 + l16;
      const unsigned short* base = &As[row * LDSK + hlf * 8];
      af[mt].f[0] = *(const F4*)(base);        // K = hlf*8 .. +7
      af[mt].f[1] = *(const F4*)(base + 16);   // K = 16 + hlf*8 .. +7
    }
#pragma unroll
    for (int nt = 0; nt < 2; ++nt) {
      int col = wn + nt * 16 + l16;
      const unsigned short* base = &Bs[col * LDSK + hlf * 8];
      bfr[nt].f[0] = *(const F4*)(base);       // K = hlf*8 .. +7
      bfr[nt].f[1] = *(const F4*)(base + 16);  // K = 16 + hlf*8 .. +7
    }
#pragma unroll
    for (int mt = 0; mt < 4; ++mt)
#pragma unroll
      for (int nt = 0; nt < 2; ++nt)
        acc[mt][nt].v = __builtin_amdgcn_wmma_f32_16x16x32_bf16(
            false, af[mt].v, false, bfr[nt].v, (short)0, acc[mt][nt].v, false, false);
  }

  const float alpha = aptr ? *aptr : aimm;
#pragma unroll
  for (int mt = 0; mt < 4; ++mt) {
    int gm0 = mBase + wm + mt * 16 + hlf * 8;
#pragma unroll
    for (int nt = 0; nt < 2; ++nt) {
      int gn = nBase + wn + nt * 16 + l16;
      if (gn >= N) continue;
      float bv = bias ? bias[gn] : 0.f;
#pragma unroll
      for (int r = 0; r < 8; ++r) {
        int gm = gm0 + r;
        if (gm >= M) continue;
        float v = alpha * acc[mt][nt].f[r] + bv;
        if (EPI == 1)      v = 0.5f * v * (1.f + erff(v * 0.70710678118f));
        else if (EPI == 2) v = 1.f / (1.f + expf(-v));
        else if (EPI == 3) v = 2.f / (1.f + expf(-v));
        Co[(long long)gm * ldc + gn] = v;
      }
    }
  }
}

// ---------------------------------------------------------------------------
// Small supporting kernels
// ---------------------------------------------------------------------------

__global__ void embed_pos_kernel(const int* __restrict__ ids, const float* __restrict__ emb,
                                 float* __restrict__ out, int total, int C, int S) {
  int idx = blockIdx.x * blockDim.x + threadIdx.x;
  if (idx >= total) return;
  int tok = idx / C, c = idx - tok * C;
  int s = tok % S;
  float ang = (float)s * expf(-logf(10000.f) * (float)(c & ~1) / (float)C);
  out[idx] = emb[(long long)ids[tok] * C + c] + ((c & 1) ? cosf(ang) : sinf(ang));
}

__global__ void ext_init_kernel(float* __restrict__ x, const float* __restrict__ mem, int total) {
  int idx = blockIdx.x * blockDim.x + threadIdx.x;
  if (idx >= total) return;
  int t = idx >> 10, j = idx & 1023;           // 2 ext slots * 512
  x[(long long)t * 3072 + 2048 + j] = mem[j];
}

__global__ __launch_bounds__(256)
void rmsnorm_kernel(const float* __restrict__ in, long long istr,
                    float* __restrict__ out, long long ostr,
                    const float* __restrict__ w, int D) {
  int row = blockIdx.x;
  const float* xr = in + (long long)row * istr;
  float* orow = out + (long long)row * ostr;
  float ss = 0.f;
  for (int d = threadIdx.x; d < D; d += 256) { float v = xr[d]; ss += v * v; }
  __shared__ float red[256];
  red[threadIdx.x] = ss; __syncthreads();
  for (int s = 128; s > 0; s >>= 1) {
    if ((int)threadIdx.x < s) red[threadIdx.x] += red[threadIdx.x + s];
    __syncthreads();
  }
  float rs = rsqrtf(red[0] / (float)D + 1e-6f);
  for (int d = threadIdx.x; d < D; d += 256) orow[d] = xr[d] * rs * w[d];
}

__global__ void sinkhorn_kernel(float* __restrict__ H, int BSn) {
  int t = blockIdx.x * blockDim.x + threadIdx.x;
  if (t >= BSn) return;
  float z[36];
#pragma unroll
  for (int k = 0; k < 36; ++k) z[k] = H[t * 36 + k] * 20.0f;   // / tau (0.05)
  for (int it = 0; it < 10; ++it) {
#pragma unroll
    for (int i = 0; i < 6; ++i) {
      float m = z[i * 6];
#pragma unroll
      for (int j = 1; j < 6; ++j) m = fmaxf(m, z[i * 6 + j]);
      float s = 0.f;
#pragma unroll
      for (int j = 0; j < 6; ++j) s += expf(z[i * 6 + j] - m);
      float lse = m + logf(s);
#pragma unroll
      for (int j = 0; j < 6; ++j) z[i * 6 + j] -= lse;
    }
#pragma unroll
    for (int j = 0; j < 6; ++j) {
      float m = z[j];
#pragma unroll
      for (int i = 1; i < 6; ++i) m = fmaxf(m, z[i * 6 + j]);
      float s = 0.f;
#pragma unroll
      for (int i = 0; i < 6; ++i) s += expf(z[i * 6 + j] - m);
      float lse = m + logf(s);
#pragma unroll
      for (int i = 0; i < 6; ++i) z[i * 6 + j] -= lse;
    }
  }
#pragma unroll
  for (int k = 0; k < 36; ++k) H[t * 36 + k] = expf(z[k]);
}

__global__ void hin_kernel(const float* __restrict__ Hpre, const float* __restrict__ xn,
                           float* __restrict__ hin, int total) {
  int idx = blockIdx.x * blockDim.x + threadIdx.x;
  if (idx >= total) return;
  int t = idx >> 9, c = idx & 511;
  const float* hp = Hpre + t * 6;
  const float* xr = xn + (long long)t * 3072 + c;
  float a = 0.f;
#pragma unroll
  for (int n = 0; n < 6; ++n) a += hp[n] * xr[n * 512];
  hin[idx] = a;
}

// softmax over causal prefix; emits normalized probabilities as bf16
__global__ __launch_bounds__(256)
void softmax_causal_kernel(float* __restrict__ P, unsigned short* __restrict__ Pb, int S) {
  long long row = blockIdx.x;
  int q = (int)(row % S);
  float* p = P + row * (long long)S;
  unsigned short* pb = Pb + row * (long long)S;
  int n = q + 1;
  __shared__ float red[256];
  float m = -1e30f;
  for (int j = threadIdx.x; j < n; j += 256) m = fmaxf(m, p[j]);
  red[threadIdx.x] = m; __syncthreads();
  for (int s = 128; s > 0; s >>= 1) {
    if ((int)threadIdx.x < s) red[threadIdx.x] = fmaxf(red[threadIdx.x], red[threadIdx.x + s]);
    __syncthreads();
  }
  m = red[0]; __syncthreads();
  float ssum = 0.f;
  for (int j = threadIdx.x; j < n; j += 256) { float e = expf(p[j] - m); p[j] = e; ssum += e; }
  red[threadIdx.x] = ssum; __syncthreads();
  for (int s = 128; s > 0; s >>= 1) {
    if ((int)threadIdx.x < s) red[threadIdx.x] += red[threadIdx.x + s];
    __syncthreads();
  }
  float inv = 1.f / red[0];
  for (int j = threadIdx.x; j < n; j += 256) pb[j] = bf_bits(p[j] * inv);
  for (int j = n + threadIdx.x; j < S; j += 256) pb[j] = 0;  // exact full-K P@V
}

__global__ void conn_post_kernel(const float* __restrict__ x, const float* __restrict__ hout,
                                 const float* __restrict__ Hres, const float* __restrict__ Hpost,
                                 const float* __restrict__ gate, const float* __restrict__ mem,
                                 float* __restrict__ xnew, int total) {
  int idx = blockIdx.x * blockDim.x + threadIdx.x;
  if (idx >= total) return;
  int t = idx >> 9, c = idx & 511;
  float xv[6];
#pragma unroll
  for (int j = 0; j < 6; ++j) xv[j] = x[(long long)t * 3072 + j * 512 + c];
  float ho = hout[idx];
  const float* hr = Hres + t * 36;
  const float* hp = Hpost + t * 6;
#pragma unroll
  for (int i = 0; i < 6; ++i) {
    float a = hp[i] * ho;
#pragma unroll
    for (int j = 0; j < 6; ++j) a += hr[i * 6 + j] * xv[j];
    if (i >= 4) {
      float g = 1.f / (1.f + expf(-gate[i - 4]));
      a = (1.f - g) * a + g * mem[(i - 4) * 512 + c];
    }
    xnew[(long long)t * 3072 + i * 512 + c] = a;
  }
}

// ---------------------------------------------------------------------------
// Host orchestration
// ---------------------------------------------------------------------------

struct GP {
  const void* A; long long lda;
  const float* B; long long ldb;
  float* C; long long ldc;
  const float* bias; const float* aptr; float aimm;
  int M, N, K, batches, nH;
  long long sAb, sAh, sBb, sBh, sCb, sCh;
};

static GP mk(const void* A, long long lda, const float* B, long long ldb,
             float* C, long long ldc, const float* bias, const float* aptr, float aimm,
             int M, int N, int K) {
  GP g{}; g.A = A; g.lda = lda; g.B = B; g.ldb = ldb; g.C = C; g.ldc = ldc;
  g.bias = bias; g.aptr = aptr; g.aimm = aimm; g.M = M; g.N = N; g.K = K;
  g.batches = 1; g.nH = 1;
  g.sAb = g.sAh = g.sBb = g.sBh = g.sCb = g.sCh = 0;
  return g;
}

static void gemm(hipStream_t st, int epi, bool bt, const GP& g) {
  dim3 grid((g.N + BN - 1) / BN, (g.M + BM - 1) / BM, g.batches);
  dim3 blk(256);
#define GCALL(E, T, F) gemm_nt_kernel<E, T, F><<<grid, blk, 0, st>>>( \
    g.A, g.lda, g.B, g.ldb, g.C, g.ldc, g.bias, g.aptr, g.aimm, \
    g.M, g.N, g.K, g.nH, g.sAb, g.sAh, g.sBb, g.sBh, g.sCb, g.sCh)
  if (bt)            { GCALL(0, true, true); }    // P@V: bf16 A + K-major V
  else if (epi == 0) { GCALL(0, false, false); }
  else if (epi == 1) { GCALL(1, false, false); }
  else if (epi == 2) { GCALL(2, false, false); }
  else               { GCALL(3, false, false); }
#undef GCALL
}

extern "C" void kernel_launch(void* const* d_in, const int* in_sizes, int n_in,
                              void* d_out, int out_size, void* d_ws, size_t ws_size,
                              hipStream_t stream) {
  (void)in_sizes; (void)n_in; (void)out_size; (void)ws_size;
  const int BSn = 2048, Cc = 512, NCn = 3072, Sn = 1024, VOC = 32000, DFF = 2048;

  auto f = [&](int i) { return (const float*)d_in[i]; };
  // JAX pytree flattening: dict keys sorted alphabetically at each level.
  const int* ids               = (const int*)d_in[0];
  const float* embedding       = f(1);
  const float* final_norm_w    = f(2);
  const float* final_proj_b    = f(3);
  const float* final_proj_w    = f(4);
  const float* initial_proj_b  = f(5);
  const float* initial_proj_w  = f(6);

  struct Conn { const float *a_post,*a_pre,*a_res,*b_post,*b_pre,*b_res,*gate,*mem,*norm_w,*phi_post,*phi_pre,*phi_res; };
  struct Layer { const float *in_b,*in_w,*out_b,*out_w; Conn c1, c2;
                 const float *ffn_b1,*ffn_b2,*ffn_w1,*ffn_w2,*norm1_w,*norm2_w; };
  Layer Ly[4];
  int p = 7;
  for (int l = 0; l < 4; ++l) {
    Layer& L = Ly[l];
    L.in_b = f(p++); L.in_w = f(p++); L.out_b = f(p++); L.out_w = f(p++);
    Conn* cs[2] = { &L.c1, &L.c2 };
    for (int ci = 0; ci < 2; ++ci) {
      Conn& c = *cs[ci];
      c.a_post = f(p++); c.a_pre = f(p++); c.a_res = f(p++);
      c.b_post = f(p++); c.b_pre = f(p++); c.b_res = f(p++);
      c.gate = f(p++); c.mem = f(p++); c.norm_w = f(p++);
      c.phi_post = f(p++); c.phi_pre = f(p++); c.phi_res = f(p++);
    }
    L.ffn_b1 = f(p++); L.ffn_b2 = f(p++); L.ffn_w1 = f(p++); L.ffn_w2 = f(p++);
    L.norm1_w = f(p++); L.norm2_w = f(p++);
  }
  const float* lm_head_b = f(p++);
  const float* lm_head_w = f(p++);

  // workspace layout (floats), ~181 MB total
  float* w    = (float*)d_ws;
  float* x    = w;  w += (size_t)BSn * NCn;        // state (B,S,6,512)
  float* xn   = w;  w += (size_t)BSn * NCn;        // normed state / next state
  float* hin  = w;  w += (size_t)BSn * Cc;         // gathered sublayer input (also embed buf)
  float* hno  = w;  w += (size_t)BSn * Cc;         // rmsnormed sublayer input
  float* qkv  = w;  w += (size_t)BSn * 3 * Cc;     // qkv (also final proj output)
  float* ao   = w;  w += (size_t)BSn * Cc;         // attention output (pre out-proj)
  float* hout = w;  w += (size_t)BSn * Cc;         // sublayer output
  float* big  = w;  w += (size_t)16 * Sn * Sn;     // scores / ffn hidden / final tok-norm
  float* Hpre = w;  w += (size_t)BSn * 6;
  float* Hpost= w;  w += (size_t)BSn * 6;
  float* Hres = w;  w += (size_t)BSn * 36;
  unsigned short* Pbf = (unsigned short*)w;        // bf16 attention probabilities
  w += (size_t)8 * Sn * Sn;                        // 16*S*S ushorts

  // 1) embedding + positional encoding -> hin (B*S, 512)
  embed_pos_kernel<<<(BSn * Cc) / 256, 256, 0, stream>>>(ids, embedding, hin, BSn * Cc, Cc, Sn);

  // 2) initial projection -> token slots of x (ldc = 3072)
  gemm(stream, 0, false, mk(hin, Cc, initial_proj_w, Cc, x, NCn,
                            initial_proj_b, nullptr, 1.f, BSn, 2048, Cc));

  // 3) ext slots <- conn1.mem of layer 0
  ext_init_kernel<<<(BSn * 1024) / 256, 256, 0, stream>>>(x, Ly[0].c1.mem, BSn * 1024);

  for (int l = 0; l < 4; ++l) {
    for (int sub = 0; sub < 2; ++sub) {
      const Conn& cn = (sub == 0) ? Ly[l].c1 : Ly[l].c2;

      // xn = rmsnorm(x, norm_w) over 3072
      rmsnorm_kernel<<<BSn, 256, 0, stream>>>(x, NCn, xn, NCn, cn.norm_w, NCn);

      // H_pre = sigmoid(a_pre * xn @ phi_pre^T + b_pre)
      gemm(stream, 2, false, mk(xn, NCn, cn.phi_pre, NCn, Hpre, 6,
                                cn.b_pre, cn.a_pre, 1.f, BSn, 6, NCn));
      // H_post = 2*sigmoid(a_post * xn @ phi_post^T + b_post)
      gemm(stream, 3, false, mk(xn, NCn, cn.phi_post, NCn, Hpost, 6,
                                cn.b_post, cn.a_post, 1.f, BSn, 6, NCn));
      // H_res raw = a_res * xn @ phi_res^T + b_res ; then Sinkhorn
      gemm(stream, 0, false, mk(xn, NCn, cn.phi_res, NCn, Hres, 36,
                                cn.b_res, cn.a_res, 1.f, BSn, 36, NCn));
      sinkhorn_kernel<<<BSn / 256, 256, 0, stream>>>(Hres, BSn);

      // h_in[t,c] = sum_n Hpre[t,n] * xn[t,n,c]
      hin_kernel<<<(BSn * Cc) / 256, 256, 0, stream>>>(Hpre, xn, hin, BSn * Cc);

      if (sub == 0) {
        // ---- MHA sublayer ----
        rmsnorm_kernel<<<BSn, 256, 0, stream>>>(hin, Cc, hno, Cc, Ly[l].norm1_w, Cc);
        gemm(stream, 0, false, mk(hno, Cc, Ly[l].in_w, Cc, qkv, 3 * Cc,
                                  Ly[l].in_b, nullptr, 1.f, BSn, 3 * Cc, Cc));
        // scores = (Q @ K^T) / 8, batched over (b,h) = 16
        {
          GP g = mk(qkv, 3 * Cc, qkv + Cc, 3 * Cc, big, Sn,
                    nullptr, nullptr, 0.125f, Sn, Sn, 64);
          g.batches = 16; g.nH = 8;
          g.sAb = (long long)Sn * 3 * Cc; g.sAh = 64;
          g.sBb = (long long)Sn * 3 * Cc; g.sBh = 64;
          g.sCb = (long long)8 * Sn * Sn; g.sCh = (long long)Sn * Sn;
          gemm(stream, 0, false, g);
        }
        softmax_causal_kernel<<<16 * Sn, 256, 0, stream>>>(big, Pbf, Sn);
        // O = P @ V  (P is bf16 -> async LDS staging; V is K-major -> BTRANS)
        {
          GP g = mk(Pbf, Sn, qkv + 2 * Cc, 3 * Cc, ao, Cc,
                    nullptr, nullptr, 1.f, Sn, 64, Sn);
          g.batches = 16; g.nH = 8;
          g.sAb = (long long)8 * Sn * Sn; g.sAh = (long long)Sn * Sn;
          g.sBb = (long long)Sn * 3 * Cc; g.sBh = 64;
          g.sCb = (long long)Sn * Cc;     g.sCh = 64;
          gemm(stream, 0, true, g);
        }
        gemm(stream, 0, false, mk(ao, Cc, Ly[l].out_w, Cc, hout, Cc,
                                  Ly[l].out_b, nullptr, 1.f, BSn, Cc, Cc));
      } else {
        // ---- FFN sublayer ----
        rmsnorm_kernel<<<BSn, 256, 0, stream>>>(hin, Cc, hno, Cc, Ly[l].norm2_w, Cc);
        gemm(stream, 1, false, mk(hno, Cc, Ly[l].ffn_w1, Cc, big, DFF,
                                  Ly[l].ffn_b1, nullptr, 1.f, BSn, DFF, Cc));   // exact GELU
        gemm(stream, 0, false, mk(big, DFF, Ly[l].ffn_w2, DFF, hout, Cc,
                                  Ly[l].ffn_b2, nullptr, 1.f, BSn, Cc, DFF));
      }

      // x_new = H_res @ x + H_post * h_out, ext slots blended with gated memory
      conn_post_kernel<<<(BSn * Cc) / 256, 256, 0, stream>>>(
          x, hout, Hres, Hpost, cn.gate, cn.mem, xn, BSn * Cc);
      float* tmp = x; x = xn; xn = tmp;
    }
  }

  // final: rmsnorm over token slots (D=2048, row stride 3072) -> big
  rmsnorm_kernel<<<BSn, 256, 0, stream>>>(x, NCn, big, 2048, final_norm_w, 2048);
  // final_proj: (2048x2048)@(512x2048)^T -> qkv (reused as (B*S,512))
  gemm(stream, 0, false, mk(big, 2048, final_proj_w, 2048, qkv, Cc,
                            final_proj_b, nullptr, 1.f, BSn, Cc, 2048));
  // lm_head: (2048x512)@(32000x512)^T -> d_out
  gemm(stream, 0, false, mk(qkv, Cc, lm_head_w, Cc, (float*)d_out, VOC,
                            lm_head_b, nullptr, 1.f, BSn, VOC, Cc));
}
